// CrossModalAttention_55405078119337
// MI455X (gfx1250) — matrix operand order repaired and checked by reading, not compiled, not run on hardware
//
#include <hip/hip_runtime.h>

// CDNA5 / gfx1250: wave32, WMMA 16x16x32 bf16 with fp32 accumulation.
typedef __attribute__((ext_vector_type(16))) __bf16 v16bf;
typedef __attribute__((ext_vector_type(8)))  float  v8f;

#define TB 128           // 4 wave32s per block, arranged 2x2
#define PITCH 48         // LDS row pitch in bf16 elems: 96B, 16B aligned, low-conflict

struct alignas(16) U4 { unsigned x, y, z, w; };
struct U8 { U4 lo, hi; };

__device__ __forceinline__ unsigned short f2bf(float f) {
  unsigned u = __builtin_bit_cast(unsigned, f);
  u += 0x7FFFu + ((u >> 16) & 1u);           // round-to-nearest-even
  return (unsigned short)(u >> 16);
}

// Build a v16bf A/B fragment from two 16-byte LDS reads (ds_load_b128).
__device__ __forceinline__ v16bf mkfrag(const unsigned short* p0, const unsigned short* p1) {
  U8 u;
  u.lo = *(const U4*)p0;
  u.hi = *(const U4*)p1;
  return __builtin_bit_cast(v16bf, u);
}

__device__ __forceinline__ v8f wmma_bf16(v16bf a, v16bf b, v8f c) {
  // D = A(16x32 bf16) * B(32x16 bf16) + C(16x16 f32)
  return __builtin_amdgcn_wmma_f32_16x16x32_bf16(false, a, false, b, (short)0, c, false, false);
}

// ---------------------------------------------------------------------------
// Kernel P: proj = bf16( X[M x D](f32) @ W[D x U](f32) + bias )
// Block tile 64x64; each wave computes 32x32 via 2x2 WMMA accumulators.
// Double-buffered LDS staging with fp32->bf16 conversion.
// ---------------------------------------------------------------------------
__global__ __launch_bounds__(TB) void proj_bf16_kernel(
    const float* __restrict__ X, const float* __restrict__ W,
    const float* __restrict__ bias, unsigned short* __restrict__ out,
    int D, int U) {
  __shared__ __align__(16) unsigned short sA[2][64 * PITCH];
  __shared__ __align__(16) unsigned short sB[2][64 * PITCH];
  const int tid  = threadIdx.x;
  const int lane = tid & 31, wave = tid >> 5;
  const int half = lane >> 4, ln16 = lane & 15;
  const int wm = wave >> 1, wn = wave & 1;
  const long long m0 = (long long)blockIdx.x * 64;
  const int n0 = blockIdx.y * 64;

  v8f acc[2][2];
  #pragma unroll
  for (int im = 0; im < 2; ++im)
    #pragma unroll
    for (int in = 0; in < 2; ++in)
      acc[im][in] = (v8f){0.f,0.f,0.f,0.f,0.f,0.f,0.f,0.f};

  auto stage = [&](int buf, int k0) {
    { // A tile 64x32 (fp32 -> bf16), 2 threads per row, 16 elems each
      const int r = tid >> 1, c = (tid & 1) * 16;
      const float* src = X + (m0 + r) * (long long)D + k0 + c;
      #pragma unroll
      for (int j = 0; j < 16; ++j) sA[buf][r * PITCH + c + j] = f2bf(src[j]);
    }
    { // B tile transposed: sB[n][kk] = W[k0+kk][n0+n], 32 k-rows x 64 n
      const int kk = tid >> 2, nc = (tid & 3) * 16;
      const float* src = W + (long long)(k0 + kk) * U + n0 + nc;
      #pragma unroll
      for (int j = 0; j < 16; ++j) sB[buf][(nc + j) * PITCH + kk] = f2bf(src[j]);
    }
  };

  stage(0, 0);
  __syncthreads();
  int buf = 0;
  for (int k0 = 0; k0 < D; k0 += 32) {
    if (k0 + 32 < D) stage(buf ^ 1, k0 + 32);
    const unsigned short* a0p = &sA[buf][(wm * 32 + ln16) * PITCH];
    const unsigned short* a1p = &sA[buf][(wm * 32 + 16 + ln16) * PITCH];
    const unsigned short* b0p = &sB[buf][(wn * 32 + ln16) * PITCH];
    const unsigned short* b1p = &sB[buf][(wn * 32 + 16 + ln16) * PITCH];
    v16bf a0 = mkfrag(a0p + half * 8, a0p + 16 + half * 8);
    v16bf a1 = mkfrag(a1p + half * 8, a1p + 16 + half * 8);
    v16bf b0 = mkfrag(b0p + half * 8, b0p + 16 + half * 8);
    v16bf b1 = mkfrag(b1p + half * 8, b1p + 16 + half * 8);
    acc[0][0] = wmma_bf16(a0, b0, acc[0][0]);
    acc[0][1] = wmma_bf16(a0, b1, acc[0][1]);
    acc[1][0] = wmma_bf16(a1, b0, acc[1][0]);
    acc[1][1] = wmma_bf16(a1, b1, acc[1][1]);
    __syncthreads();
    buf ^= 1;
  }
  #pragma unroll
  for (int im = 0; im < 2; ++im) {
    const long long mb = m0 + wm * 32 + im * 16 + half * 8;  // C/D: m = 8*half + v
    #pragma unroll
    for (int in = 0; in < 2; ++in) {
      const int n = n0 + wn * 32 + in * 16 + ln16;
      const float bi = bias[n];
      #pragma unroll
      for (int v = 0; v < 8; ++v)
        out[(mb + v) * U + n] = f2bf(acc[im][in][v] + bi);
    }
  }
}

// ---------------------------------------------------------------------------
// Kernel G: batched C(f32) = A(bf16) @ B(bf16), fp32 accumulate.
//  transA=0: A stored [M][K] (ldA);  transA=1: A stored [K][M] (ldA)
//  transB=0: B stored [K][N] (ldB);  transB=1: B stored [N][K] (ldB)
// Block tile 64x64, wave tile 32x32 (4 WMMA / K-step), double-buffered LDS.
// ---------------------------------------------------------------------------
__global__ __launch_bounds__(TB) void gemm_bf16f32_kernel(
    const unsigned short* __restrict__ A, long long ldA, long long aBatch, int transA,
    const unsigned short* __restrict__ Bm, long long ldB, long long bBatch, int transB,
    float* __restrict__ C, long long ldC, long long cBatch, int K) {
  __shared__ __align__(16) unsigned short sA[2][64 * PITCH];
  __shared__ __align__(16) unsigned short sB[2][64 * PITCH];
  const int tid  = threadIdx.x;
  const int lane = tid & 31, wave = tid >> 5;
  const int half = lane >> 4, ln16 = lane & 15;
  const int wm = wave >> 1, wn = wave & 1;
  const long long m0 = (long long)blockIdx.x * 64;
  const long long n0 = (long long)blockIdx.y * 64;
  const unsigned short* Ab = A  + aBatch * blockIdx.z;
  const unsigned short* Bb = Bm + bBatch * blockIdx.z;
  float* Cb = C + cBatch * blockIdx.z;

  v8f acc[2][2];
  #pragma unroll
  for (int im = 0; im < 2; ++im)
    #pragma unroll
    for (int in = 0; in < 2; ++in)
      acc[im][in] = (v8f){0.f,0.f,0.f,0.f,0.f,0.f,0.f,0.f};

  auto stage = [&](int buf, int k0) {
    if (!transA) {            // direct copy, 2x b128 per thread
      const int r = tid >> 1, c = (tid & 1) * 16;
      const unsigned short* gp = Ab + (m0 + r) * ldA + k0 + c;
      const U4* src = (const U4*)gp;
      *(U4*)(&sA[buf][r * PITCH + c])     = src[0];
      *(U4*)(&sA[buf][r * PITCH + c + 8]) = src[1];
      if (k0 + 32 < K)        // global_prefetch_b8 the tile after next
        __builtin_prefetch((const void*)(gp + 32), 0, 1);
    } else {                  // transpose-stage: sA[m][kk] = A[k0+kk][m0+m]
      const int kk = tid >> 2, mc = (tid & 3) * 16;
      const U4* src = (const U4*)(Ab + (long long)(k0 + kk) * ldA + m0 + mc);
      union { U4 q[2]; unsigned short s[16]; } u;
      u.q[0] = src[0]; u.q[1] = src[1];
      #pragma unroll
      for (int j = 0; j < 16; ++j) sA[buf][(mc + j) * PITCH + kk] = u.s[j];
    }
    if (transB) {             // B stored [N][K]: direct copy into sB[n][k]
      const int n = tid >> 1, kc = (tid & 1) * 16;
      const U4* src = (const U4*)(Bb + (n0 + n) * ldB + k0 + kc);
      *(U4*)(&sB[buf][n * PITCH + kc])     = src[0];
      *(U4*)(&sB[buf][n * PITCH + kc + 8]) = src[1];
    } else {                  // B stored [K][N]: transpose-stage
      const int kk = tid >> 2, nc = (tid & 3) * 16;
      const U4* src = (const U4*)(Bb + (long long)(k0 + kk) * ldB + n0 + nc);
      union { U4 q[2]; unsigned short s[16]; } u;
      u.q[0] = src[0]; u.q[1] = src[1];
      #pragma unroll
      for (int j = 0; j < 16; ++j) sB[buf][(nc + j) * PITCH + kk] = u.s[j];
    }
  };

  stage(0, 0);
  __syncthreads();
  int buf = 0;
  for (int k0 = 0; k0 < K; k0 += 32) {
    if (k0 + 32 < K) stage(buf ^ 1, k0 + 32);
    const unsigned short* a0p = &sA[buf][(wm * 32 + ln16) * PITCH];
    const unsigned short* a1p = &sA[buf][(wm * 32 + 16 + ln16) * PITCH];
    const unsigned short* b0p = &sB[buf][(wn * 32 + ln16) * PITCH];
    const unsigned short* b1p = &sB[buf][(wn * 32 + 16 + ln16) * PITCH];
    v16bf a0 = mkfrag(a0p + half * 8, a0p + 16 + half * 8);
    v16bf a1 = mkfrag(a1p + half * 8, a1p + 16 + half * 8);
    v16bf b0 = mkfrag(b0p + half * 8, b0p + 16 + half * 8);
    v16bf b1 = mkfrag(b1p + half * 8, b1p + 16 + half * 8);
    acc[0][0] = wmma_bf16(a0, b0, acc[0][0]);
    acc[0][1] = wmma_bf16(a0, b1, acc[0][1]);
    acc[1][0] = wmma_bf16(a1, b0, acc[1][0]);
    acc[1][1] = wmma_bf16(a1, b1, acc[1][1]);
    __syncthreads();
    buf ^= 1;
  }
  #pragma unroll
  for (int im = 0; im < 2; ++im) {
    const long long mb = m0 + wm * 32 + im * 16 + half * 8;  // C/D: m = 8*half + v
    #pragma unroll
    for (int in = 0; in < 2; ++in) {
      const long long n = n0 + wn * 32 + in * 16 + ln16;
      #pragma unroll
      for (int v = 0; v < 8; ++v)
        Cb[(mb + v) * ldC + n] = acc[im][in][v];
    }
  }
}

// ---------------------------------------------------------------------------
// Softmax over each length-T row of fp32 scores; writes bf16 attn overlaid
// into the first 2*T bytes of the same row (row pitch stays T fp32 = 2T bf16).
// One 256-thread block per row, 8 elements/thread (T = 2048).
// ---------------------------------------------------------------------------
__global__ __launch_bounds__(256) void softmax_bf16_inplace_kernel(
    float* __restrict__ scores, int T) {
  const long long row = blockIdx.x;
  float* rp = scores + row * (long long)T;
  const int t = threadIdx.x;
  float v[8];
  float m = -3.402823466e38f;
  #pragma unroll
  for (int i = 0; i < 8; ++i) { v[i] = rp[t + (i << 8)]; m = fmaxf(m, v[i]); }
  __shared__ float red[256];
  red[t] = m; __syncthreads();
  for (int s = 128; s > 0; s >>= 1) {
    if (t < s) red[t] = fmaxf(red[t], red[t + s]);
    __syncthreads();
  }
  m = red[0]; __syncthreads();
  float sum = 0.f;
  #pragma unroll
  for (int i = 0; i < 8; ++i) { v[i] = __expf(v[i] - m); sum += v[i]; }
  red[t] = sum; __syncthreads();
  for (int s = 128; s > 0; s >>= 1) {
    if (t < s) red[t] += red[t + s];
    __syncthreads();
  }
  const float inv = 1.0f / red[0];
  // All fp32 reads of this row happened before the first barrier; safe to
  // overwrite the front half of the row with bf16 now.
  unsigned short* op = (unsigned short*)rp;
  #pragma unroll
  for (int i = 0; i < 8; ++i) op[t + (i << 8)] = f2bf(v[i] * inv);
}

// ---------------------------------------------------------------------------
extern "C" void kernel_launch(void* const* d_in, const int* in_sizes, int n_in,
                              void* d_out, int out_size, void* d_ws, size_t ws_size,
                              hipStream_t stream) {
  const int Bt = 8, T = 2048, DE = 512, DI = 768, U = 256;
  const float* eeg   = (const float*)d_in[0];
  const float* image = (const float*)d_in[1];
  const float* W_e   = (const float*)d_in[2];
  const float* b_e   = (const float*)d_in[3];
  const float* W_i   = (const float*)d_in[4];
  const float* b_i   = (const float*)d_in[5];

  const long long M = (long long)Bt * T;                 // 16384 total rows
  // ws layout: [proj_e bf16: 8MB][proj_i bf16: 8MB][scores f32 / attn bf16 overlay: 128MB]
  unsigned short* proj_e = (unsigned short*)d_ws;
  unsigned short* proj_i = proj_e + (size_t)M * U;
  float* scores = (float*)((char*)d_ws + 2 * (size_t)M * U * sizeof(unsigned short));
  float* out_e = (float*)d_out;                          // attended_eeg  [B,T,U]
  float* out_i = out_e + (size_t)M * U;                  // attended_image[B,T,U]

  dim3 blk(TB);

  // 1-2) projections to shared U space (bf16)
  proj_bf16_kernel<<<dim3((unsigned)(M / 64), U / 64, 1), blk, 0, stream>>>(
      eeg, W_e, b_e, proj_e, DE, U);
  proj_bf16_kernel<<<dim3((unsigned)(M / 64), U / 64, 1), blk, 0, stream>>>(
      image, W_i, b_i, proj_i, DI, U);

  // 3) scores[b] = proj_e[b] @ proj_i[b]^T   (B stored [N][K] -> transB=1)
  gemm_bf16f32_kernel<<<dim3(T / 64, T / 64, Bt), blk, 0, stream>>>(
      proj_e, U, (long long)T * U, 0,
      proj_i, U, (long long)T * U, 1,
      scores, T, (long long)T * T, U);

  // 4) row softmax; attn written as bf16 overlay (row pitch = 2*T bf16 elems)
  softmax_bf16_inplace_kernel<<<dim3((unsigned)(Bt * T)), dim3(256), 0, stream>>>(scores, T);

  // 5) attended_eeg[b] = attn[b] @ proj_e[b]
  gemm_bf16f32_kernel<<<dim3(T / 64, U / 64, Bt), blk, 0, stream>>>(
      (const unsigned short*)scores, 2 * T, 2LL * T * T, 0,
      proj_e, U, (long long)T * U, 0,
      out_e, U, (long long)T * U, T);

  // 6) attended_image[b] = attn[b]^T @ proj_i[b]  (A stored [K][M] -> transA=1)
  gemm_bf16f32_kernel<<<dim3(T / 64, U / 64, Bt), blk, 0, stream>>>(
      (const unsigned short*)scores, 2 * T, 2LL * T * T, 1,
      proj_i, U, (long long)T * U, 0,
      out_i, U, (long long)T * U, T);
}
